// PointerDecoder_45569603011105
// MI455X (gfx1250) — matrix operand order repaired and checked by reading, not compiled, not run on hardware
//
#include <hip/hip_runtime.h>
#include <hip/hip_bf16.h>

// ---------------------------------------------------------------------------
// Types for CDNA5 WMMA (gfx1250, wave32)
// ---------------------------------------------------------------------------
typedef __attribute__((ext_vector_type(16))) __bf16 v16bf;
typedef __attribute__((ext_vector_type(8)))  float  v8f;

struct FragBF {
    union {
        v16bf v;
        uint4 q[2];
        unsigned short u[16];
    };
};

__device__ __forceinline__ unsigned short f32_to_bf16_rne(float f) {
    unsigned int u = __float_as_uint(f);
    unsigned int r = (u + 0x7FFFu + ((u >> 16) & 1u)) >> 16;
    return (unsigned short)r;
}

__device__ __forceinline__ float sigmoidf_(float x) {
    return 1.0f / (1.0f + __expf(-x));
}

// ---------------------------------------------------------------------------
// One 16x16 f32 output tile, A[m0..m0+15][0..K) (row-major bf16, ld = lda)
// times W^T where W is row-major [N][K] bf16 (ld = ldw), accumulated over K.
// K must be a multiple of 32.  Per-lane loads follow the ISA fragment layout:
//   A (16-bit 16x32): lane group g = lane>>4 holds K = {g*8..g*8+7, 16+g*8..}
//   B (16-bit 32x16): lane holds col N = lane&15, K = {g*8..g*8+7, 16+g*8..}
// ---------------------------------------------------------------------------
__device__ __forceinline__ v8f wmma_tile_bf16(const unsigned short* __restrict__ A, int lda, int m0,
                                              const unsigned short* __restrict__ W, int ldw, int n0,
                                              int K, int lane) {
    v8f acc = {};
    const int cl = lane & 15;
    const int g  = lane >> 4;
    const unsigned short* arow = A + (size_t)(m0 + cl) * lda;
    const unsigned short* wrow = W + (size_t)(n0 + cl) * ldw;
#pragma unroll 4
    for (int k0 = 0; k0 < K; k0 += 32) {
        FragBF a, b;
        const unsigned short* ab = arow + k0 + g * 8;
        a.q[0] = *(const uint4*)(ab);
        a.q[1] = *(const uint4*)(ab + 16);
        const unsigned short* bb = wrow + k0 + g * 8;
        b.q[0] = *(const uint4*)(bb);
        b.q[1] = *(const uint4*)(bb + 16);
        acc = __builtin_amdgcn_wmma_f32_16x16x32_bf16(
            false, a.v, false, b.v, (short)0, acc, false, false);
    }
    return acc;
}

// ---------------------------------------------------------------------------
// Generic GEMM: out[M][N] = act( A(bf16)[M][K] * W(bf16)[N][K]^T + bias[N] )
// grid = (M/64, N/64), block = 256 (8 waves, 2 16x16 tiles per wave)
// act: 0 = none, 1 = tanh
// ---------------------------------------------------------------------------
__global__ __launch_bounds__(256)
void gemm_bias(const unsigned short* __restrict__ A, int lda,
               const unsigned short* __restrict__ W, int ldw,
               const float* __restrict__ bias,
               float* __restrict__ out, int ldo,
               int K, int act) {
    const int tm   = blockIdx.x * 64;
    const int tn   = blockIdx.y * 64;
    const int wave = threadIdx.x >> 5;
    const int lane = threadIdx.x & 31;
    const int cl   = lane & 15;
    const int g    = lane >> 4;
#pragma unroll
    for (int tix = wave * 2; tix < wave * 2 + 2; ++tix) {
        const int mi = tix >> 2;
        const int ni = tix & 3;
        const int m0 = tm + mi * 16;
        const int n0 = tn + ni * 16;
        v8f acc = wmma_tile_bf16(A, lda, m0, W, ldw, n0, K, lane);
        const int col = n0 + cl;
        const float bs = bias[col];
#pragma unroll
        for (int i = 0; i < 8; ++i) {
            const int r = m0 + g * 8 + i;
            float val = acc[i] + bs;
            if (act == 1) val = tanhf(val);
            out[(size_t)r * ldo + col] = val;
        }
    }
}

// ---------------------------------------------------------------------------
// Fused LSTM step: one block per 16 h-columns (grid = 32 blocks).
// Computes all 4 gate strips (64x16, K=1024) with WMMA, then the cell.
// ---------------------------------------------------------------------------
__global__ __launch_bounds__(256)
void lstm_step(const unsigned short* __restrict__ xh,   // [64][1024] bf16
               const unsigned short* __restrict__ wg,   // [2048][1024] bf16 ([Wih|Whh])
               const float* __restrict__ b_ih, const float* __restrict__ b_hh,
               float* __restrict__ c,                   // [64][512] in/out
               float* __restrict__ h_lstm,              // [64][512]
               unsigned short* __restrict__ hl_bf,      // [64][512] bf16
               unsigned short* __restrict__ ah_bf) {    // [64][1024] bf16 ([attn|h_lstm])
    __shared__ float gl[4][64][16];   // 16 KB
    const int j    = blockIdx.x;      // h columns [j*16, j*16+16)
    const int wave = threadIdx.x >> 5;
    const int lane = threadIdx.x & 31;
    const int cl   = lane & 15;
    const int g    = lane >> 4;
#pragma unroll
    for (int tix = wave * 2; tix < wave * 2 + 2; ++tix) {
        const int gate = tix >> 2;      // 0=i 1=f 2=g 3=o
        const int m0   = (tix & 3) * 16;
        const int n0   = gate * 512 + j * 16;
        v8f acc = wmma_tile_bf16(xh, 1024, m0, wg, 1024, n0, 1024, lane);
#pragma unroll
        for (int i = 0; i < 8; ++i) gl[gate][m0 + g * 8 + i][cl] = acc[i];
    }
    __syncthreads();
    for (int idx = threadIdx.x; idx < 64 * 16; idx += 256) {
        const int b  = idx >> 4;
        const int cc = idx & 15;
        const int hc = j * 16 + cc;                 // global h column
        const float gi = gl[0][b][cc] + b_ih[hc]          + b_hh[hc];
        const float gf = gl[1][b][cc] + b_ih[512 + hc]    + b_hh[512 + hc];
        const float gg = gl[2][b][cc] + b_ih[1024 + hc]   + b_hh[1024 + hc];
        const float go = gl[3][b][cc] + b_ih[1536 + hc]   + b_hh[1536 + hc];
        const size_t o = (size_t)b * 512 + hc;
        const float ct = sigmoidf_(gf) * c[o] + sigmoidf_(gi) * tanhf(gg);
        const float hl = sigmoidf_(go) * tanhf(ct);
        c[o]      = ct;
        h_lstm[o] = hl;
        const unsigned short bh = f32_to_bf16_rne(hl);
        hl_bf[o]                    = bh;
        ah_bf[(size_t)b * 1024 + 512 + hc] = bh;
    }
}

// ---------------------------------------------------------------------------
// Attention + pointer selection. One block (256 thr) per batch row.
// ---------------------------------------------------------------------------
__global__ __launch_bounds__(256)
void attn_step(const float* __restrict__ ctx,       // [64][128][512]
               const float* __restrict__ inp,       // [64][512]
               const float* __restrict__ v,         // [512]
               float* __restrict__ mask,            // [64][128] in/out
               const float* __restrict__ max_len,   // [64]
               const float* __restrict__ inputs,    // [64][128][512]
               float* __restrict__ attn,            // [64][512]
               unsigned short* __restrict__ ah_bf,  // [64][1024]
               float* __restrict__ x,               // [64][512] next LSTM input
               float* __restrict__ out_scores,      // [64][128][128]
               float* __restrict__ out_ptr,         // [64][128]
               int t_step) {
    __shared__ float sP[256];
    __shared__ float sE[128];
    __shared__ float sAlpha[128];
    __shared__ float red[128];
    __shared__ int   sSel;
    const int b   = blockIdx.x;
    const int tid = threadIdx.x;
    const float* ctxb  = ctx + (size_t)b * 128 * 512;
    const float* inpb  = inp + (size_t)b * 512;
    float*       maskb = mask + (size_t)b * 128;

    // e[t] = sum_h v[h] * tanh(ctx[b][t][h] + inp[b][h]); 2 threads per t
    {
        const int t = tid >> 1, half = tid & 1;
        const float* row = ctxb + (size_t)t * 512 + half * 256;
        const float* ip  = inpb + half * 256;
        const float* vp  = v + half * 256;
        float s = 0.f;
        for (int hh = 0; hh < 256; ++hh) s += vp[hh] * tanhf(row[hh] + ip[hh]);
        sP[tid] = s;
    }
    __syncthreads();
    if (tid < 128) {
        const float e  = sP[2 * tid] + sP[2 * tid + 1];
        const float sc = (maskb[tid] > 0.f) ? e : -1e9f;
        sE[tid]  = sc;
        red[tid] = sc;
        out_scores[(size_t)b * 128 * 128 + (size_t)t_step * 128 + tid] = sc;
    }
    __syncthreads();
    // softmax over 128
    for (int s = 64; s > 0; s >>= 1) {
        if (tid < s) red[tid] = fmaxf(red[tid], red[tid + s]);
        __syncthreads();
    }
    const float mx = red[0];
    __syncthreads();
    if (tid < 128) {
        const float ex = __expf(sE[tid] - mx);
        sAlpha[tid] = ex;
        red[tid]    = ex;
    }
    __syncthreads();
    for (int s = 64; s > 0; s >>= 1) {
        if (tid < s) red[tid] += red[tid + s];
        __syncthreads();
    }
    const float denom = red[0];
    __syncthreads();
    if (tid < 128) sAlpha[tid] = sAlpha[tid] / denom;
    __syncthreads();
    // pointer = first argmax of alpha*mask (jnp.argmax first-max semantics)
    if (tid == 0) {
        float best = -1.f;
        int   bi   = 0;
        for (int t = 0; t < 128; ++t) {
            const float p = sAlpha[t] * maskb[t];
            if (p > best) { best = p; bi = t; }
        }
        int sel = (max_len[b] > (float)t_step) ? bi : t_step;
        sSel = sel;
        maskb[sel] = 0.f;   // repeat=False: remove chosen position
        out_ptr[(size_t)b * 128 + t_step] = (float)sel;
    }
    __syncthreads();
    const int sel = sSel;
    // attn[b][g] = sum_t alpha[t] * ctx[b][t][g]
    for (int g2 = tid; g2 < 512; g2 += 256) {
        float a = 0.f;
#pragma unroll 8
        for (int t = 0; t < 128; ++t) a += sAlpha[t] * ctxb[(size_t)t * 512 + g2];
        attn[(size_t)b * 512 + g2]  = a;
        ah_bf[(size_t)b * 1024 + g2] = f32_to_bf16_rne(a);
    }
    // x_next = inputs[b][sel][:]
    const float* srcx = inputs + ((size_t)b * 128 + sel) * 512;
    for (int d2 = tid; d2 < 512; d2 += 256) x[(size_t)b * 512 + d2] = srcx[d2];
}

// ---------------------------------------------------------------------------
// Small helper kernels
// ---------------------------------------------------------------------------
__global__ void cvt_bf16(const float* __restrict__ src, unsigned short* __restrict__ dst, int n) {
    for (int i = blockIdx.x * blockDim.x + threadIdx.x; i < n; i += gridDim.x * blockDim.x)
        dst[i] = f32_to_bf16_rne(src[i]);
}

__global__ void build_wg(const float* __restrict__ wih, const float* __restrict__ whh,
                         unsigned short* __restrict__ wg) {
    const int n = 2048 * 1024;
    for (int i = blockIdx.x * blockDim.x + threadIdx.x; i < n; i += gridDim.x * blockDim.x) {
        const int r = i >> 10, k = i & 1023;
        const float val = (k < 512) ? wih[(size_t)r * 512 + k] : whh[(size_t)r * 512 + (k - 512)];
        wg[i] = f32_to_bf16_rne(val);
    }
}

__global__ void pack_xh(const float* __restrict__ x, const float* __restrict__ h,
                        unsigned short* __restrict__ xh) {
    const int n = 64 * 1024;
    for (int i = blockIdx.x * blockDim.x + threadIdx.x; i < n; i += gridDim.x * blockDim.x) {
        const int b = i >> 10, k = i & 1023;
        const float val = (k < 512) ? x[(size_t)b * 512 + k] : h[(size_t)b * 512 + (k - 512)];
        xh[i] = f32_to_bf16_rne(val);
    }
}

__global__ void init_state(const float* __restrict__ h0, const float* __restrict__ c0,
                           const float* __restrict__ cmask, const float* __restrict__ input0,
                           float* __restrict__ h, float* __restrict__ c, float* __restrict__ mask,
                           float* __restrict__ x, float* __restrict__ mlen) {
    const int idx = blockIdx.x * blockDim.x + threadIdx.x;
    if (idx < 64 * 512) {
        h[idx] = h0[idx];
        c[idx] = c0[idx];
        x[idx] = input0[idx & 511];
    }
    if (idx < 64 * 128) mask[idx] = cmask[idx];
    if (idx < 64) {
        float s = 0.f;
        for (int t = 0; t < 128; ++t) s += cmask[idx * 128 + t];
        mlen[idx] = s;
    }
}

__global__ void finalize(const float* __restrict__ h, const float* __restrict__ c,
                         float* __restrict__ hf, float* __restrict__ cf) {
    const int n = 64 * 512;
    for (int i = blockIdx.x * blockDim.x + threadIdx.x; i < n; i += gridDim.x * blockDim.x) {
        hf[i] = h[i];
        cf[i] = c[i];
    }
}

// ---------------------------------------------------------------------------
// Host-side orchestration (graph-capture safe: stream-ordered launches only)
// ---------------------------------------------------------------------------
extern "C" void kernel_launch(void* const* d_in, const int* in_sizes, int n_in,
                              void* d_out, int out_size, void* d_ws, size_t ws_size,
                              hipStream_t stream) {
    const float* inputs = (const float*)d_in[0];   // [64,128,512]
    const float* h0     = (const float*)d_in[1];   // [64,512]
    const float* c0     = (const float*)d_in[2];   // [64,512]
    const float* cand   = (const float*)d_in[3];   // [64,128,512]
    const float* cmask  = (const float*)d_in[4];   // [64,128]
    const float* input0 = (const float*)d_in[5];   // [512]
    const float* W_ih   = (const float*)d_in[6];   // [2048,512]
    const float* b_ih   = (const float*)d_in[7];   // [2048]
    const float* W_hh   = (const float*)d_in[8];   // [2048,512]
    const float* b_hh   = (const float*)d_in[9];   // [2048]
    const float* W_out  = (const float*)d_in[10];  // [512,1024]
    const float* b_out  = (const float*)d_in[11];  // [512]
    const float* W_inp  = (const float*)d_in[12];  // [512,512]
    const float* b_inp  = (const float*)d_in[13];  // [512]
    const float* W_ctx  = (const float*)d_in[14];  // [512,512]
    const float* b_ctx  = (const float*)d_in[15];  // [512]
    const float* v      = (const float*)d_in[16];  // [512]

    // d_out layout: outputs [64][128][128] | pointers [64][128] | h_f [64][512] | c_f [64][512]
    float* out_scores = (float*)d_out;
    float* out_ptr    = out_scores + (size_t)64 * 128 * 128;
    float* out_hf     = out_ptr + (size_t)64 * 128;
    float* out_cf     = out_hf + (size_t)64 * 512;

    // workspace carve-out (256B aligned)
    char*  ws  = (char*)d_ws;
    size_t off = 0;
    auto take = [&](size_t bytes) -> char* {
        char* p = ws + off;
        off += (bytes + 255) & ~(size_t)255;
        return p;
    };
    unsigned short* wg    = (unsigned short*)take((size_t)2048 * 1024 * 2);
    unsigned short* winp  = (unsigned short*)take((size_t)512 * 512 * 2);
    unsigned short* woutw = (unsigned short*)take((size_t)512 * 1024 * 2);
    unsigned short* wctx  = (unsigned short*)take((size_t)512 * 512 * 2);
    unsigned short* candb = (unsigned short*)take((size_t)8192 * 512 * 2);
    float*          ctx   = (float*)take((size_t)8192 * 512 * 4);
    unsigned short* xh    = (unsigned short*)take((size_t)64 * 1024 * 2);
    unsigned short* hlb   = (unsigned short*)take((size_t)64 * 512 * 2);
    unsigned short* ahb   = (unsigned short*)take((size_t)64 * 1024 * 2);
    float*          h     = (float*)take((size_t)64 * 512 * 4);
    float*          c     = (float*)take((size_t)64 * 512 * 4);
    float*          x     = (float*)take((size_t)64 * 512 * 4);
    float*          hl    = (float*)take((size_t)64 * 512 * 4);
    float*          inp   = (float*)take((size_t)64 * 512 * 4);
    float*          attn  = (float*)take((size_t)64 * 512 * 4);
    float*          mask  = (float*)take((size_t)64 * 128 * 4);
    float*          mlen  = (float*)take((size_t)64 * 4);
    (void)ws_size; (void)in_sizes; (void)n_in; (void)out_size;

    // one-time conversions (re-run every call: deterministic, input-only)
    cvt_bf16<<<1024, 256, 0, stream>>>(W_inp, winp, 512 * 512);
    cvt_bf16<<<2048, 256, 0, stream>>>(W_out, woutw, 512 * 1024);
    cvt_bf16<<<1024, 256, 0, stream>>>(W_ctx, wctx, 512 * 512);
    cvt_bf16<<<8192, 256, 0, stream>>>(cand, candb, 8192 * 512);
    build_wg<<<8192, 256, 0, stream>>>(W_ih, W_hh, wg);

    // ctx_lin = candidate @ W_ctx^T + b_ctx   (M=8192, N=512, K=512)
    gemm_bias<<<dim3(128, 8), 256, 0, stream>>>(candb, 512, wctx, 512, b_ctx, ctx, 512, 512, 0);

    init_state<<<128, 256, 0, stream>>>(h0, c0, cmask, input0, h, c, mask, x, mlen);

    for (int t = 0; t < 128; ++t) {
        pack_xh<<<256, 256, 0, stream>>>(x, h, xh);
        lstm_step<<<32, 256, 0, stream>>>(xh, wg, b_ih, b_hh, c, hl, hlb, ahb);
        // inp = h_lstm @ W_inp^T + b_inp   (M=64, N=512, K=512)
        gemm_bias<<<dim3(1, 8), 256, 0, stream>>>(hlb, 512, winp, 512, b_inp, inp, 512, 512, 0);
        attn_step<<<64, 256, 0, stream>>>(ctx, inp, v, mask, mlen, inputs,
                                          attn, ahb, x, out_scores, out_ptr, t);
        // h = tanh([attn, h_lstm] @ W_out^T + b_out)   (M=64, N=512, K=1024)
        gemm_bias<<<dim3(1, 8), 256, 0, stream>>>(ahb, 1024, woutw, 1024, b_out, h, 512, 1024, 1);
    }
    finalize<<<256, 256, 0, stream>>>(h, c, out_hf, out_cf);
}
